// MultiHeadSelfAttention_74509092651425
// MI455X (gfx1250) — matrix operand order repaired and checked by reading, not compiled, not run on hardware
//
#include <hip/hip_runtime.h>
#include <math.h>

// ---------------------------------------------------------------------------
// MI455X / gfx1250 multi-head self-attention (width-attention, conv-1x1 proj)
// All GEMM stages run on V_WMMA_F32_16X16X32_BF16 (fp32 accumulation).
// Two-pass softmax (row stats + recompute) avoids materializing [B,H,W,W].
// All strided (k-major) operands are staged through LDS with transpose-on-
// store, so every WMMA fragment load is a 16B ds_load_b128 / global_load_b128.
// ---------------------------------------------------------------------------

typedef __bf16 bf16_t;
typedef __attribute__((ext_vector_type(16))) __bf16        v16bf;
typedef __attribute__((ext_vector_type(8)))  float         v8f;
typedef __attribute__((ext_vector_type(4)))  unsigned int  v4u;

#define B_    4
#define H_    4
#define C_    512
#define W_    2048
#define D_    64
#define MALL  2560   // 256 (q) + 256 (k) + 2048 (v) rows

#define WMMA_BF16(a, b, c) \
  __builtin_amdgcn_wmma_f32_16x16x32_bf16(false, (a), false, (b), (short)0, (c), false, false)

union FragU { v16bf v; v4u q[2]; };
union Pack8 { v4u q; bf16_t h[8]; };

__device__ __forceinline__ v8f v8f_zero() {
  v8f z;
#pragma unroll
  for (int i = 0; i < 8; ++i) z[i] = 0.0f;
  return z;
}

// ---- fragment loads (all 16B-vectorized) ----------------------------------
// A-matrix 16x32 bf16 layout: m = lane&15, k = (lane>>4)*8 + {0..7, 16..23}
// B-matrix 32x16 bf16 layout: n = lane&15, k = (lane>>4)*8 + {0..7, 16..23}

// A fragment, element (m,k) at p[m*ld + k] (row-major; global Wall / V)
__device__ __forceinline__ v16bf load_a_row(const bf16_t* p, int ld, int lane) {
  const int m  = lane & 15;
  const int kb = (lane >> 4) * 8;
  FragU u;
  u.q[0] = *(const v4u*)(p + (size_t)m * ld + kb);
  u.q[1] = *(const v4u*)(p + (size_t)m * ld + kb + 16);
  return u.v;
}
// A fragment from transposed LDS image: element (m,k) at base[m*ld + k]
__device__ __forceinline__ v16bf load_a_ldsT(const bf16_t* base, int ld, int kk, int lane) {
  const int m  = lane & 15;
  const int kb = kk + (lane >> 4) * 8;
  FragU u;
  u.q[0] = *(const v4u*)(base + m * ld + kb);
  u.q[1] = *(const v4u*)(base + m * ld + kb + 16);
  return u.v;
}
// B fragment from transposed LDS image: element (k,n) at base[n*ld + k]
__device__ __forceinline__ v16bf load_b_ldsT(const bf16_t* base, int ld, int kk, int lane) {
  const int n  = lane & 15;
  const int kb = kk + (lane >> 4) * 8;
  FragU u;
  u.q[0] = *(const v4u*)(base + n * ld + kb);
  u.q[1] = *(const v4u*)(base + n * ld + kb + 16);
  return u.v;
}

// --------------------------- pack kernels ----------------------------------

__global__ void pack_weights_bf16(const float* __restrict__ Wq, const float* __restrict__ bq,
                                  const float* __restrict__ Wk, const float* __restrict__ bk,
                                  const float* __restrict__ Wv, const float* __restrict__ bv,
                                  bf16_t* __restrict__ Wall, float* __restrict__ ball) {
  int idx = blockIdx.x * blockDim.x + threadIdx.x;
  if (idx >= MALL * C_) return;
  int m = idx >> 9;   // / 512
  int c = idx & 511;
  float w, bias;
  if (m < 256) {                       // Q rows
    int h = m >> 6, d = m & 63;
    w = Wq[(size_t)(h * D_ + d) * C_ + c];
    bias = bq[h * D_ + d];
  } else if (m < 512) {                // K rows
    int mm = m - 256;
    int h = mm >> 6, d = mm & 63;
    w = Wk[(size_t)(h * D_ + d) * C_ + c];
    bias = bk[h * D_ + d];
  } else {                             // V rows
    int mm = m - 512;
    int h = mm >> 9, e = mm & 511;
    w = Wv[(size_t)(h * C_ + e) * C_ + c];
    bias = bv[h * C_ + e];
  }
  Wall[idx] = (bf16_t)w;
  if (c == 0) ball[m] = bias;
}

__global__ void pack_x_bf16(const float* __restrict__ x, bf16_t* __restrict__ xbf) {
  int idx = blockIdx.x * blockDim.x + threadIdx.x;
  if (idx < B_ * C_ * W_) xbf[idx] = (bf16_t)x[idx];
}

// --------------------------- QKV projection --------------------------------
// out[b, m, w] = sum_c Wall[m,c] * x[b,c,w] + ball[m], bf16 out.
// grid: (N/64, M/128, B), 256 threads (8 waves). Wave w: rows m0..m0+15, 4 n-tiles.
// x-tile [32k x 64n] is staged transposed into LDS (xsT[n][k], ld=40).
__global__ void qkv_proj_wmma(const bf16_t* __restrict__ Wall, const float* __restrict__ ball,
                              const bf16_t* __restrict__ xbf, bf16_t* __restrict__ qkv) {
  const int nt = blockIdx.x;
  const int mt = blockIdx.y;
  const int b  = blockIdx.z;
  const int tid  = threadIdx.x;
  const int wave = tid >> 5;
  const int lane = tid & 31;
  const int m0 = mt * 128 + wave * 16;
  const int n0 = nt * 64;
  const bf16_t* xb = xbf + (size_t)b * C_ * W_;

  __shared__ bf16_t xsT[64 * 40];   // [n_local][k], padded stride 40

  // staging map: thread -> (k row, 8-wide n group); 8 lanes cover one 128B row
  const int sk = tid >> 3;          // 0..31
  const int sn = (tid & 7) * 8;     // 0..56

  v8f acc[4];
#pragma unroll
  for (int t = 0; t < 4; ++t) acc[t] = v8f_zero();

  for (int kk = 0; kk < C_; kk += 32) {
    __syncthreads();   // protect previous tile's reads
    {
      const bf16_t* src = xb + (size_t)(kk + sk) * W_ + n0 + sn;
      __builtin_prefetch((const void*)(src + (size_t)32 * W_), 0, 3);
      Pack8 w; w.q = *(const v4u*)src;
#pragma unroll
      for (int e = 0; e < 8; ++e) xsT[(sn + e) * 40 + sk] = w.h[e];
    }
    __syncthreads();

    v16bf a = load_a_row(Wall + (size_t)m0 * C_ + kk, C_, lane);
#pragma unroll
    for (int t = 0; t < 4; ++t) {
      v16bf bb = load_b_ldsT(xsT + (t * 16) * 40, 40, 0, lane);
      acc[t] = WMMA_BF16(a, bb, acc[t]);
    }
  }

  const int hi = lane >> 4, nl = lane & 15;
#pragma unroll
  for (int t = 0; t < 4; ++t) {
#pragma unroll
    for (int r = 0; r < 8; ++r) {
      const int m = m0 + hi * 8 + r;
      const int n = n0 + t * 16 + nl;
      const float v = acc[t][r] + ball[m];
      qkv[((size_t)b * MALL + m) * W_ + n] = (bf16_t)v;
    }
  }
}

// --------------------------- softmax row stats -----------------------------
// stats[bh*W + i] = (rowmax_i, rowsumexp_i) of S[i,:] = scale * Q^T K.
// grid: (W/32, B*H), 256 threads. Block owns 32 i-rows; waves sweep j in 128-wide
// strips. Q^T staged once (QsT[i][d]); K strip staged per step (KsT[j][d]).
__global__ void attn_row_stats_wmma(const bf16_t* __restrict__ qkv, float2* __restrict__ stats) {
  const int it = blockIdx.x;
  const int bh = blockIdx.y;
  const int b = bh >> 2, h = bh & 3;
  const int tid  = threadIdx.x;
  const int wave = tid >> 5;
  const int lane = tid & 31;
  const int hi = lane >> 4;
  const int i0 = it * 32;
  const bf16_t* Q = qkv + ((size_t)b * MALL + h * D_) * W_;
  const bf16_t* K = qkv + ((size_t)b * MALL + 256 + h * D_) * W_;

  __shared__ bf16_t QsT[32 * 72];    // [i_local][d], ld=72
  __shared__ bf16_t KsT[128 * 72];   // [j_local][d], ld=72
  __shared__ float sm[8][32];
  __shared__ float sl[8][32];

  // stage Q^T once: thread -> (d row, 8-wide i group)
  {
    const int d  = tid >> 2;         // 0..63
    const int ig = (tid & 3) * 8;    // 0..24
    Pack8 w; w.q = *(const v4u*)(Q + (size_t)d * W_ + i0 + ig);
#pragma unroll
    for (int e = 0; e < 8; ++e) QsT[(ig + e) * 72 + d] = w.h[e];
  }

  float rmax[2][8], rsum[2][8];
#pragma unroll
  for (int im = 0; im < 2; ++im)
#pragma unroll
    for (int r = 0; r < 8; ++r) { rmax[im][r] = -INFINITY; rsum[im][r] = 0.0f; }

  for (int jj = 0; jj < W_ / 128; ++jj) {
    __syncthreads();   // previous strip consumed (and QsT visible on jj==0)
    // stage 128-wide K strip: 1024 16B chunks, 4 per thread, coalesced per row
#pragma unroll
    for (int c = 0; c < 4; ++c) {
      const int id = c * 256 + tid;
      const int d  = id >> 4;          // 0..63
      const int jg = (id & 15) * 8;    // 0..120
      Pack8 w; w.q = *(const v4u*)(K + (size_t)d * W_ + jj * 128 + jg);
#pragma unroll
      for (int e = 0; e < 8; ++e) KsT[(jg + e) * 72 + d] = w.h[e];
    }
    __syncthreads();

#pragma unroll
    for (int im = 0; im < 2; ++im) {
      v8f s = v8f_zero();
#pragma unroll
      for (int kk = 0; kk < D_; kk += 32) {
        v16bf a  = load_a_ldsT(QsT + (im * 16) * 72, 72, kk, lane);
        v16bf bb = load_b_ldsT(KsT + (wave * 16) * 72, 72, kk, lane);
        s = WMMA_BF16(a, bb, s);
      }
#pragma unroll
      for (int r = 0; r < 8; ++r) {
        const float v  = s[r] * 0.125f;       // scale = D^-0.5
        const float nm = fmaxf(rmax[im][r], v);
        rsum[im][r] = rsum[im][r] * __expf(rmax[im][r] - nm) + __expf(v - nm);
        rmax[im][r] = nm;
      }
    }
  }

  // reduce across the 16 lanes (columns) of each half-wave
#pragma unroll
  for (int im = 0; im < 2; ++im) {
#pragma unroll
    for (int r = 0; r < 8; ++r) {
      float m = rmax[im][r], l = rsum[im][r];
      for (int mask = 1; mask < 16; mask <<= 1) {
        const float om = __shfl_xor(m, mask, 32);
        const float ol = __shfl_xor(l, mask, 32);
        const float nm = fmaxf(m, om);
        l = l * __expf(m - nm) + ol * __expf(om - nm);
        m = nm;
      }
      rmax[im][r] = m; rsum[im][r] = l;
    }
  }
  if ((lane & 15) == 0) {
#pragma unroll
    for (int im = 0; im < 2; ++im)
#pragma unroll
      for (int r = 0; r < 8; ++r) {
        sm[wave][im * 16 + hi * 8 + r] = rmax[im][r];
        sl[wave][im * 16 + hi * 8 + r] = rsum[im][r];
      }
  }
  __syncthreads();
  if (threadIdx.x < 32) {
    const int il = threadIdx.x;
    float m = -INFINITY, l = 0.0f;
    for (int wv = 0; wv < 8; ++wv) {
      const float om = sm[wv][il], ol = sl[wv][il];
      const float nm = fmaxf(m, om);
      l = l * __expf(m - nm) + ol * __expf(om - nm);
      m = nm;
    }
    stats[(size_t)bh * W_ + i0 + il] = make_float2(m, l);
  }
}

// --------------------------- context + epilogue ----------------------------
// out[b,c,j] = 2*x[b,c,j] + sum_h sum_i V_h[c,i] * softmax_row(S_h)[i,j]
// grid: (W/32, B), 256 threads. Block owns a 32-wide j-tile; loops h & 32-row
// i-chunks. K^T staged per h; Q^T staged per chunk; prob tile lAT stored
// j-major so S-phase lanes emit one packed 16B store each.
__global__ void attn_ctx_out_wmma(const bf16_t* __restrict__ qkv,
                                  const float2* __restrict__ stats,
                                  const float* __restrict__ x, float* __restrict__ out) {
  const int jt = blockIdx.x;
  const int b  = blockIdx.y;
  const int tid  = threadIdx.x;
  const int wave = tid >> 5;
  const int lane = tid & 31;
  const int hi = lane >> 4, nl = lane & 15;
  const int j0 = jt * 32;

  __shared__ bf16_t KsT[32 * 72];   // [j_local][d]
  __shared__ bf16_t QsT[32 * 72];   // [i_local][d]
  __shared__ bf16_t lAT[32 * 40];   // [j_local][i_local], padded stride 40

  v8f acc[4][2];
#pragma unroll
  for (int mt = 0; mt < 4; ++mt) { acc[mt][0] = v8f_zero(); acc[mt][1] = v8f_zero(); }

  const int sd = tid >> 2;          // 0..63 (staging row)
  const int sg = (tid & 3) * 8;     // 0..24 (staging 8-wide group)

  for (int h = 0; h < H_; ++h) {
    const bf16_t* Q = qkv + ((size_t)b * MALL + h * D_) * W_;
    const bf16_t* K = qkv + ((size_t)b * MALL + 256 + h * D_) * W_;
    const bf16_t* V = qkv + ((size_t)b * MALL + 512 + h * C_) * W_;
    const float2* st = stats + (size_t)(b * H_ + h) * W_;

    // stage K^T for this head (fixed j-tile)
    {
      Pack8 w; w.q = *(const v4u*)(K + (size_t)sd * W_ + j0 + sg);
#pragma unroll
      for (int e = 0; e < 8; ++e) KsT[(sg + e) * 72 + sd] = w.h[e];
    }
    __syncthreads();

    for (int ic = 0; ic < W_ / 32; ++ic) {
      const int i0 = ic * 32;
      // stage Q^T for this i-chunk
      {
        const bf16_t* src = Q + (size_t)sd * W_ + i0 + sg;
        __builtin_prefetch((const void*)(src + 32), 0, 3);
        Pack8 w; w.q = *(const v4u*)src;
#pragma unroll
        for (int e = 0; e < 8; ++e) QsT[(sg + e) * 72 + sd] = w.h[e];
      }
      __syncthreads();

      if (wave < 4) {  // uniform per-wave branch: EXEC all-1s inside
        const int im = wave & 1, jn = wave >> 1;
        v8f s = v8f_zero();
#pragma unroll
        for (int kk = 0; kk < D_; kk += 32) {
          v16bf a  = load_a_ldsT(QsT + (im * 16) * 72, 72, kk, lane);
          v16bf bb = load_b_ldsT(KsT + (jn * 16) * 72, 72, kk, lane);
          s = WMMA_BF16(a, bb, s);
        }
        Pack8 w;
#pragma unroll
        for (int r = 0; r < 8; ++r) {
          const int irow = i0 + im * 16 + hi * 8 + r;
          const float2 ml = st[irow];
          w.h[r] = (bf16_t)(__expf(s[r] * 0.125f - ml.x) / ml.y);
        }
        *(v4u*)(lAT + (jn * 16 + nl) * 40 + im * 16 + hi * 8) = w.q;
      }
      __syncthreads();

      v16bf b0 = load_b_ldsT(lAT, 40, 0, lane);
      v16bf b1 = load_b_ldsT(lAT + 16 * 40, 40, 0, lane);
#pragma unroll
      for (int mt = 0; mt < 4; ++mt) {
        const int c0 = (wave * 4 + mt) * 16;
        v16bf a = load_a_row(V + (size_t)c0 * W_ + i0, W_, lane);
        acc[mt][0] = WMMA_BF16(a, b0, acc[mt][0]);
        acc[mt][1] = WMMA_BF16(a, b1, acc[mt][1]);
      }
      __syncthreads();   // before next chunk restages QsT / lAT
    }
  }

#pragma unroll
  for (int mt = 0; mt < 4; ++mt) {
#pragma unroll
    for (int jn = 0; jn < 2; ++jn) {
#pragma unroll
      for (int r = 0; r < 8; ++r) {
        const int c = (wave * 4 + mt) * 16 + hi * 8 + r;
        const int j = j0 + jn * 16 + nl;
        const size_t idx = ((size_t)b * C_ + c) * W_ + j;
        out[idx] = 2.0f * x[idx] + acc[mt][jn][r];
      }
    }
  }
}

// --------------------------- launch ----------------------------------------

extern "C" void kernel_launch(void* const* d_in, const int* in_sizes, int n_in,
                              void* d_out, int out_size, void* d_ws, size_t ws_size,
                              hipStream_t stream) {
  const float* x  = (const float*)d_in[0];
  const float* Wq = (const float*)d_in[1];
  const float* bq = (const float*)d_in[2];
  const float* Wk = (const float*)d_in[3];
  const float* bk = (const float*)d_in[4];
  const float* Wv = (const float*)d_in[5];
  const float* bv = (const float*)d_in[6];
  float* out = (float*)d_out;

  // workspace carve-up (256B aligned)
  char* ws = (char*)d_ws;
  size_t off = 0;
  auto take = [&](size_t bytes) { char* p = ws + off; off = (off + bytes + 255) & ~(size_t)255; return p; };
  bf16_t* Wall  = (bf16_t*)take((size_t)MALL * C_ * sizeof(bf16_t));          // 2.6 MB
  float*  ball  = (float*) take((size_t)MALL * sizeof(float));                // 10 KB
  bf16_t* xbf   = (bf16_t*)take((size_t)B_ * C_ * W_ * sizeof(bf16_t));       // 8.4 MB
  bf16_t* qkv   = (bf16_t*)take((size_t)B_ * MALL * W_ * sizeof(bf16_t));     // 41.9 MB
  float2* stats = (float2*)take((size_t)B_ * H_ * W_ * sizeof(float2));       // 0.26 MB
  (void)ws_size;

  {
    const int n = MALL * C_;
    pack_weights_bf16<<<(n + 255) / 256, 256, 0, stream>>>(Wq, bq, Wk, bk, Wv, bv, Wall, ball);
  }
  {
    const int n = B_ * C_ * W_;
    pack_x_bf16<<<(n + 255) / 256, 256, 0, stream>>>(x, xbf);
  }
  {
    dim3 grid(W_ / 64, MALL / 128, B_);   // (32, 20, 4)
    qkv_proj_wmma<<<grid, 256, 0, stream>>>(Wall, ball, xbf, qkv);
  }
  {
    dim3 grid(W_ / 32, B_ * H_);          // (64, 16)
    attn_row_stats_wmma<<<grid, 256, 0, stream>>>(qkv, stats);
  }
  {
    dim3 grid(W_ / 32, B_);               // (64, 4)
    attn_ctx_out_wmma<<<grid, 256, 0, stream>>>(qkv, stats, x, out);
  }
}
